// RegionFeatureEmbedding_26834955665548
// MI455X (gfx1250) — matrix operand order repaired and checked by reading, not compiled, not run on hardware
//
#include <hip/hip_runtime.h>
#include <hip/hip_bf16.h>
#include <cstdint>

// ---------------------------------------------------------------------------
// Problem constants: B=64, M=512, H=1024, NH=8, HD=128
// ---------------------------------------------------------------------------
constexpr int CB  = 64;
constexpr int CM  = 512;
constexpr int CH  = 1024;
constexpr int CNH = 8;
constexpr int CHD = 128;

typedef __attribute__((ext_vector_type(16))) __bf16 v16bf;
typedef __attribute__((ext_vector_type(8)))  float  v8f;

__device__ __forceinline__ __bf16 f2bf(float f) {
    unsigned u = __builtin_bit_cast(unsigned, f);
    unsigned r = u + 0x7FFFu + ((u >> 16) & 1u);   // round-to-nearest-even
    return __builtin_bit_cast(__bf16, (unsigned short)(r >> 16));
}

__device__ __forceinline__ v8f wmma_bf16(const v16bf& a, const v16bf& b, const v8f& c) {
    return __builtin_amdgcn_wmma_f32_16x16x32_bf16(
        false, a, false, b, (short)0, c, false, false);
}

// ---------------------------------------------------------------------------
// CDNA5 async global->LDS copy (16B per lane), guarded with safe fallback.
// Builtin payload type is a 4 x i32 GCC vector (per hipcc diagnostic).
// ---------------------------------------------------------------------------
#if defined(__has_builtin)
#if __has_builtin(__builtin_amdgcn_global_load_async_to_lds_b128)
#define USE_ASYNC_COPY 1
#endif
#endif
#ifndef USE_ASYNC_COPY
#define USE_ASYNC_COPY 0
#endif

typedef __attribute__((__vector_size__(4 * sizeof(int)))) int async_v4i;

__device__ __forceinline__ void copy16_g2l(const void* g, void* l) {
#if USE_ASYNC_COPY
    __builtin_amdgcn_global_load_async_to_lds_b128(
        (__attribute__((address_space(1))) async_v4i*)(uintptr_t)g,
        (__attribute__((address_space(3))) async_v4i*)(uintptr_t)l,
        0, 0);
#else
    *(uint4*)l = *(const uint4*)g;
#endif
}

template<int N>
__device__ __forceinline__ void wait_async() {
#if USE_ASYNC_COPY
#if __has_builtin(__builtin_amdgcn_s_wait_asynccnt)
    __builtin_amdgcn_s_wait_asynccnt((short)N);
#else
    asm volatile("s_wait_asynccnt 0x0" ::: "memory");
#endif
#endif
}

// ---------------------------------------------------------------------------
// mean(conv_w), mean(conv_b) -> sc[0], sc[1]
// ---------------------------------------------------------------------------
__global__ __launch_bounds__(256)
void mean2_kernel(const float* __restrict__ w, const float* __restrict__ b,
                  int n, float* __restrict__ out) {
    __shared__ float sw[256], sb[256];
    float aw = 0.f, ab = 0.f;
    for (int i = threadIdx.x; i < n; i += 256) { aw += w[i]; ab += b[i]; }
    sw[threadIdx.x] = aw; sb[threadIdx.x] = ab;
    __syncthreads();
    for (int s = 128; s > 0; s >>= 1) {
        if ((int)threadIdx.x < s) {
            sw[threadIdx.x] += sw[threadIdx.x + s];
            sb[threadIdx.x] += sb[threadIdx.x + s];
        }
        __syncthreads();
    }
    if (threadIdx.x == 0) { out[0] = sw[0] / n; out[1] = sb[0] / n; }
}

// y = x*(1+mean_w) + mean_b, emitted directly as bf16 (GEMM1 A-operand)
__global__ __launch_bounds__(256)
void affine_kernel(const float* __restrict__ x, const float* __restrict__ sc,
                   __bf16* __restrict__ y, size_t n) {
    float a = 1.f + sc[0], b = sc[1];
    for (size_t i = (size_t)blockIdx.x * blockDim.x + threadIdx.x; i < n;
         i += (size_t)gridDim.x * blockDim.x)
        y[i] = f2bf(fmaf(x[i], a, b));
}

// wt[col][k] = bf16(w[k][col])  (one-time weight convert + transpose)
__global__ __launch_bounds__(256)
void cvt_w_kernel(const float* __restrict__ w, __bf16* __restrict__ wt,
                  int K, int cols) {
    size_t n = (size_t)K * cols;
    for (size_t i = (size_t)blockIdx.x * 256 + threadIdx.x; i < n;
         i += (size_t)gridDim.x * 256) {
        size_t col = i / K, k = i % K;
        wt[i] = f2bf(w[k * cols + col]);
    }
}

// ---------------------------------------------------------------------------
// bf16 WMMA GEMM: C[rows,cols] = op(A[rows,K] @ W[K,cols] + bias)
// A: bf16 row-major. Wt: bf16 TRANSPOSED [cols][K].
// OUTM: 0 = f32 out (+opt residual), 1 = bf16 out (+opt relu),
//       2 = bf16 out stored transposed per (batch,head): [B, H, M] (V operand)
// 128x128 tile / 256 threads / 8 waves (2x4), BK=32, async double-buffered LDS.
// ---------------------------------------------------------------------------
template<int RELU, int RES, int OUTM>
__global__ __launch_bounds__(256)
void gemm_wmma_kernel(const __bf16* __restrict__ A, const __bf16* __restrict__ Wt,
                      const float* __restrict__ bias, const float* __restrict__ R,
                      float* __restrict__ Cf, __bf16* __restrict__ Cb,
                      int K, int cols) {
    constexpr int BM = 128, BN = 128, BK = 32, PAD = 8;
    __shared__ __bf16 a_sm[2][BM][BK + PAD];
    __shared__ __bf16 b_sm[2][BN][BK + PAD];

    const int tid  = threadIdx.x;
    const int lane = tid & 31;
    const int wid  = tid >> 5;
    const int half = lane >> 4;
    const int lm   = lane & 15;
    const int wm   = (wid & 1) * 64;
    const int wn   = (wid >> 1) * 32;
    const int bm   = blockIdx.y * BM;
    const int bn   = blockIdx.x * BN;

    // staging: thread owns row (tid>>1), k-halves sk..sk+15 (two 16B chunks)
    const int srow = tid >> 1;
    const int sk   = (tid & 1) * 16;
    const __bf16* ag = A  + (size_t)(bm + srow) * K + sk;
    const __bf16* bg = Wt + (size_t)(bn + srow) * K + sk;

    const v8f z8 = {0.f,0.f,0.f,0.f,0.f,0.f,0.f,0.f};
    v8f acc[4][2];
    #pragma unroll
    for (int i = 0; i < 4; ++i)
        #pragma unroll
        for (int j = 0; j < 2; ++j) acc[i][j] = z8;

    auto issue = [&](int buf, int kb) {
        copy16_g2l(ag + kb,     &a_sm[buf][srow][sk]);
        copy16_g2l(ag + kb + 8, &a_sm[buf][srow][sk + 8]);
        copy16_g2l(bg + kb,     &b_sm[buf][srow][sk]);
        copy16_g2l(bg + kb + 8, &b_sm[buf][srow][sk + 8]);
    };

    const int nk = K / BK;
    issue(0, 0);
    for (int t = 0; t < nk; ++t) {
        const int cur = t & 1;
        if (t + 1 < nk) { issue(cur ^ 1, (t + 1) * BK); wait_async<4>(); }
        else            { wait_async<0>(); }
        __syncthreads();

        v16bf bfr[2];
        #pragma unroll
        for (int ni = 0; ni < 2; ++ni)
            #pragma unroll
            for (int i = 0; i < 16; ++i)
                bfr[ni][i] = b_sm[cur][wn + ni*16 + lm][half*16 + i];

        #pragma unroll
        for (int mi = 0; mi < 4; ++mi) {
            v16bf af;
            #pragma unroll
            for (int j = 0; j < 8; ++j)
                #pragma unroll
                for (int h = 0; h < 2; ++h) {
                    int k = half * 16 + j * 2 + h;
                    af[j*2 + h] = a_sm[cur][wm + mi*16 + lm][k];
                }
            #pragma unroll
            for (int ni = 0; ni < 2; ++ni)
                acc[mi][ni] = wmma_bf16(af, bfr[ni], acc[mi][ni]);
        }
        __syncthreads();
    }

    #pragma unroll
    for (int mi = 0; mi < 4; ++mi)
        #pragma unroll
        for (int ni = 0; ni < 2; ++ni) {
            int col = bn + wn + ni*16 + lm;
            float bv = bias[col];
            #pragma unroll
            for (int r = 0; r < 8; ++r) {
                int row = bm + wm + mi*16 + half*8 + r;
                float v = acc[mi][ni][r] + bv;
                if (RELU) v = fmaxf(v, 0.f);
                if (OUTM == 0) {
                    if (RES) v += R[(size_t)row * cols + col];
                    Cf[(size_t)row * cols + col] = v;
                } else if (OUTM == 1) {
                    Cb[(size_t)row * cols + col] = f2bf(v);
                } else {  // transposed V layout: [B][col][m]
                    int bb = row / CM, m = row % CM;
                    Cb[((size_t)bb * cols + col) * CM + m] = f2bf(v);
                }
            }
        }
}

// ---------------------------------------------------------------------------
// LayerNorm over H=1024; bf16 output (feeds q/k/v GEMMs)
// ---------------------------------------------------------------------------
__global__ __launch_bounds__(256)
void layernorm_kernel(const float* __restrict__ X, const float* __restrict__ g,
                      const float* __restrict__ b, __bf16* __restrict__ Y) {
    __shared__ float red[256];
    const int tid = threadIdx.x;
    const float* xr = X + (size_t)blockIdx.x * CH;
    __bf16*      yr = Y + (size_t)blockIdx.x * CH;

    float s = 0.f;
    for (int i = tid; i < CH; i += 256) s += xr[i];
    red[tid] = s; __syncthreads();
    for (int st = 128; st > 0; st >>= 1) {
        if (tid < st) red[tid] += red[tid + st];
        __syncthreads();
    }
    float mu = red[0] / CH;
    __syncthreads();

    float vs = 0.f;
    for (int i = tid; i < CH; i += 256) { float d = xr[i] - mu; vs += d * d; }
    red[tid] = vs; __syncthreads();
    for (int st = 128; st > 0; st >>= 1) {
        if (tid < st) red[tid] += red[tid + st];
        __syncthreads();
    }
    float inv = rsqrtf(red[0] / CH + 1e-5f);

    for (int i = tid; i < CH; i += 256)
        yr[i] = f2bf((xr[i] - mu) * inv * g[i] + b[i]);
}

// ---------------------------------------------------------------------------
// Attention: 2-wave block owns 16 query rows of one (b,h).
// Q,K: bf16 [B,M,H].  Vt: bf16 [B, H, M] (per-head transposed).  O: bf16 [B,M,H].
// scores = QK^T/sqrt(HD), diag zeroed, softmax, ctx = P @ V.
// ---------------------------------------------------------------------------
__global__ __launch_bounds__(64)
void attn_kernel(const __bf16* __restrict__ Q, const __bf16* __restrict__ Kb,
                 const __bf16* __restrict__ Vt, __bf16* __restrict__ O) {
    __shared__ __bf16 q_sm[16][CHD + 8];
    // K tile [64][CHD+8] and V^T tile [CHD][64+8] alias the same storage
    __shared__ __align__(16) unsigned char kvraw[(size_t)CHD * (64 + 8) * 2];
    auto kv_sm = (__bf16 (*)[CHD + 8]) kvraw;   // [64][136]
    auto v_smT = (__bf16 (*)[64 + 8])  kvraw;   // [128][72]
    __shared__ float s_sm[16][CM];
    __shared__ float red_mx[16][4];
    __shared__ float red_sm[16][4];

    const int tid   = threadIdx.x;
    const int lane  = tid & 31;
    const int wid   = tid >> 5;
    const int half  = lane >> 4;
    const int lm    = lane & 15;
    const int qbase = blockIdx.x * 16;
    const int head  = blockIdx.y;
    const int batch = blockIdx.z;
    const size_t rowbase = (size_t)batch * CM;

    // ---- stage Q tile (16 x 128 bf16) ----
    {
        int r = tid >> 2, c0 = (tid & 3) * 32;
        const __bf16* qp = Q + (rowbase + qbase + r) * CH + head * CHD + c0;
        #pragma unroll
        for (int j = 0; j < 4; ++j)
            copy16_g2l(qp + j * 8, &q_sm[r][c0 + j * 8]);
    }
    wait_async<0>();
    __syncthreads();

    // ---- persistent Q fragments (4 K-slices of 32) ----
    v16bf af[4];
    #pragma unroll
    for (int ks = 0; ks < 4; ++ks)
        #pragma unroll
        for (int j = 0; j < 8; ++j)
            #pragma unroll
            for (int h = 0; h < 2; ++h) {
                int k = half * 16 + j * 2 + h;
                af[ks][j*2 + h] = q_sm[lm][ks*32 + k];
            }

    const float scale = 0.08838834764831845f;   // 1/sqrt(128)

    // ---- scores ----
    for (int kb = 0; kb < CM; kb += 64) {
        __syncthreads();
        const __bf16* kp = Kb + (rowbase + kb + tid) * CH + head * CHD;
        #pragma unroll
        for (int j = 0; j < 16; ++j)
            copy16_g2l(kp + j * 8, &kv_sm[tid][j * 8]);
        wait_async<0>();
        __syncthreads();

        #pragma unroll
        for (int c = 0; c < 2; ++c) {
            int ct = wid + c * 2;
            v8f acc = {0.f,0.f,0.f,0.f,0.f,0.f,0.f,0.f};
            #pragma unroll
            for (int ks = 0; ks < 4; ++ks) {
                v16bf bfr;
                #pragma unroll
                for (int i = 0; i < 16; ++i)
                    bfr[i] = kv_sm[ct*16 + lm][ks*32 + half*16 + i];
                acc = wmma_bf16(af[ks], bfr, acc);
            }
            #pragma unroll
            for (int r = 0; r < 8; ++r)
                s_sm[half*8 + r][kb + ct*16 + lm] = acc[r] * scale;
        }
    }
    __syncthreads();

    // ---- diag mask + softmax ----
    {
        int r = tid & 15, seg = tid >> 4;
        int base = seg * 128;
        int grow = qbase + r;
        float mx = -1e30f;
        for (int j = 0; j < 128; ++j) {
            int c = base + j;
            float s = s_sm[r][c];
            if (c == grow) { s = 0.f; s_sm[r][c] = 0.f; }
            mx = fmaxf(mx, s);
        }
        red_mx[r][seg] = mx;
        __syncthreads();
        mx = fmaxf(fmaxf(red_mx[r][0], red_mx[r][1]),
                   fmaxf(red_mx[r][2], red_mx[r][3]));
        float sum = 0.f;
        for (int j = 0; j < 128; ++j) {
            float e = __expf(s_sm[r][base + j] - mx);
            s_sm[r][base + j] = e;
            sum += e;
        }
        red_sm[r][seg] = sum;
        __syncthreads();
        sum = red_sm[r][0] + red_sm[r][1] + red_sm[r][2] + red_sm[r][3];
        float inv = 1.f / sum;
        for (int j = 0; j < 128; ++j) s_sm[r][base + j] *= inv;
    }
    __syncthreads();

    // ---- ctx = P @ V (V^T tiles: [d][key], contiguous fragments) ----
    const v8f z8 = {0.f,0.f,0.f,0.f,0.f,0.f,0.f,0.f};
    v8f cacc[4] = {z8, z8, z8, z8};
    const __bf16* vbase = Vt + ((size_t)batch * CH + head * CHD) * CM;
    for (int kb = 0; kb < CM; kb += 64) {
        __syncthreads();
        #pragma unroll
        for (int dd = 0; dd < 2; ++dd) {
            int d = tid * 2 + dd;
            const __bf16* vp = vbase + (size_t)d * CM + kb;
            #pragma unroll
            for (int j = 0; j < 8; ++j)
                copy16_g2l(vp + j * 8, &v_smT[d][j * 8]);
        }
        wait_async<0>();
        __syncthreads();

        #pragma unroll
        for (int ks = 0; ks < 2; ++ks) {
            v16bf pf;
            #pragma unroll
            for (int j = 0; j < 8; ++j)
                #pragma unroll
                for (int h = 0; h < 2; ++h) {
                    int k = half * 16 + j * 2 + h;
                    pf[j*2 + h] = f2bf(s_sm[lm][kb + ks*32 + k]);
                }
            #pragma unroll
            for (int nt = 0; nt < 4; ++nt) {
                int n0 = (wid * 4 + nt) * 16;
                v16bf bfr;
                #pragma unroll
                for (int i = 0; i < 16; ++i)
                    bfr[i] = v_smT[n0 + lm][ks*32 + half*16 + i];
                cacc[nt] = wmma_bf16(pf, bfr, cacc[nt]);
            }
        }
    }

    __bf16* op = O + (rowbase + qbase) * CH + head * CHD;
    #pragma unroll
    for (int nt = 0; nt < 4; ++nt) {
        int n0 = (wid * 4 + nt) * 16 + lm;
        #pragma unroll
        for (int r = 0; r < 8; ++r)
            op[(size_t)(half*8 + r) * CH + n0] = f2bf(cacc[nt][r]);
    }
}

// ---------------------------------------------------------------------------
// Launcher
// ---------------------------------------------------------------------------
extern "C" void kernel_launch(void* const* d_in, const int* in_sizes, int n_in,
                              void* d_out, int out_size, void* d_ws, size_t ws_size,
                              hipStream_t stream) {
    const float* x      = (const float*)d_in[0];
    const float* conv_w = (const float*)d_in[1];
    const float* conv_b = (const float*)d_in[2];
    const float* w1     = (const float*)d_in[3];
    const float* b1     = (const float*)d_in[4];
    const float* w2     = (const float*)d_in[5];
    const float* b2     = (const float*)d_in[6];
    const float* ln_g   = (const float*)d_in[7];
    const float* ln_b   = (const float*)d_in[8];
    const float* wq     = (const float*)d_in[9];
    const float* bq     = (const float*)d_in[10];
    const float* wk     = (const float*)d_in[11];
    const float* bk     = (const float*)d_in[12];
    const float* wv     = (const float*)d_in[13];
    const float* bv     = (const float*)d_in[14];
    const float* wo     = (const float*)d_in[15];
    const float* bo     = (const float*)d_in[16];
    const float* wp     = (const float*)d_in[17];
    const float* bp     = (const float*)d_in[18];

    const size_t SZ_X = (size_t)CB * CM * CM;
    const size_t SZ_H = (size_t)CB * CM * CH;
    const int    ROWS = CB * CM;

    char* ws = (char*)d_ws;
    size_t off = 0;
    auto alloc = [&](size_t bytes) { char* p = ws + off; off += (bytes + 255) & ~(size_t)255; return p; };

    float*  sc  = (float*) alloc(256);
    __bf16* w1t = (__bf16*)alloc((size_t)CM * CH * 2);    // [1024][512]
    __bf16* w2t = (__bf16*)alloc((size_t)CH * CH * 2);
    __bf16* wqt = (__bf16*)alloc((size_t)CH * CH * 2);
    __bf16* wkt = (__bf16*)alloc((size_t)CH * CH * 2);
    __bf16* wvt = (__bf16*)alloc((size_t)CH * CH * 2);
    __bf16* wot = (__bf16*)alloc((size_t)CH * CH * 2);
    __bf16* wpt = (__bf16*)alloc((size_t)CH * CH * 2);
    __bf16* xab = (__bf16*)alloc(SZ_X * 2);
    __bf16* h1b = (__bf16*)alloc(SZ_H * 2);
    float*  xr  = (float*) alloc(SZ_H * 4);
    __bf16* xnb = (__bf16*)alloc(SZ_H * 2);
    __bf16* qb  = (__bf16*)alloc(SZ_H * 2);
    __bf16* kbf = (__bf16*)alloc(SZ_H * 2);
    __bf16* vtb = (__bf16*)alloc(SZ_H * 2);
    __bf16* ctxb = h1b;   // h1 dead after GEMM2
    __bf16* t1b  = xnb;   // xn dead after q/k/v projections
    float*  out  = (float*)d_out;

    // 0) one-time weight convert+transpose (bf16, [cols][K])
    cvt_w_kernel<<<1024, 256, 0, stream>>>(w1, w1t, CM, CH);
    cvt_w_kernel<<<1024, 256, 0, stream>>>(w2, w2t, CH, CH);
    cvt_w_kernel<<<1024, 256, 0, stream>>>(wq, wqt, CH, CH);
    cvt_w_kernel<<<1024, 256, 0, stream>>>(wk, wkt, CH, CH);
    cvt_w_kernel<<<1024, 256, 0, stream>>>(wv, wvt, CH, CH);
    cvt_w_kernel<<<1024, 256, 0, stream>>>(wo, wot, CH, CH);
    cvt_w_kernel<<<1024, 256, 0, stream>>>(wp, wpt, CH, CH);

    // 1) conv reduction + elementwise affine (bf16 out)
    mean2_kernel<<<1, 256, 0, stream>>>(conv_w, conv_b, CH, sc);
    affine_kernel<<<4096, 256, 0, stream>>>(x, sc, xab, SZ_X);

    // 2) MLP
    gemm_wmma_kernel<1,0,1><<<dim3(CH/128, ROWS/128), 256, 0, stream>>>(
        xab, w1t, b1, nullptr, nullptr, h1b, CM, CH);
    gemm_wmma_kernel<0,0,0><<<dim3(CH/128, ROWS/128), 256, 0, stream>>>(
        h1b, w2t, b2, nullptr, xr, nullptr, CH, CH);

    // 3) LayerNorm (bf16 out)
    layernorm_kernel<<<ROWS, 256, 0, stream>>>(xr, ln_g, ln_b, xnb);

    // 4) Q/K/V projections (V stored transposed per head: [B,H,M])
    gemm_wmma_kernel<0,0,1><<<dim3(CH/128, ROWS/128), 256, 0, stream>>>(
        xnb, wqt, bq, nullptr, nullptr, qb, CH, CH);
    gemm_wmma_kernel<0,0,1><<<dim3(CH/128, ROWS/128), 256, 0, stream>>>(
        xnb, wkt, bk, nullptr, nullptr, kbf, CH, CH);
    gemm_wmma_kernel<0,0,2><<<dim3(CH/128, ROWS/128), 256, 0, stream>>>(
        xnb, wvt, bv, nullptr, nullptr, vtb, CH, CH);

    // 5) attention
    attn_kernel<<<dim3(CM/16, CNH, CB), 64, 0, stream>>>(qb, kbf, vtb, ctxb);

    // 6) output projections + residual
    gemm_wmma_kernel<0,0,1><<<dim3(CH/128, ROWS/128), 256, 0, stream>>>(
        ctxb, wot, bo, nullptr, nullptr, t1b, CH, CH);
    gemm_wmma_kernel<0,1,0><<<dim3(CH/128, ROWS/128), 256, 0, stream>>>(
        t1b, wpt, bp, xr, out, nullptr, CH, CH);
}
